// MoE_48808008352179
// MI455X (gfx1250) — matrix operand order repaired and checked by reading, not compiled, not run on hardware
//
#include <hip/hip_runtime.h>
#include <hip/hip_bf16.h>
#include <stdint.h>

// Problem constants (static per reference)
#define T_TOK  4096     // B*S = 2*2048
#define DMODEL 1024
#define NEXP   8
#define DFFN   4096
#define CAP    512      // max(4, ceil(4096/8 * 1.0))

typedef __attribute__((ext_vector_type(16))) __bf16 v16bf;
typedef __attribute__((ext_vector_type(8)))  __bf16 v8bf;
typedef __attribute__((ext_vector_type(8)))  float  v8f;

static __device__ __forceinline__ uint16_t bfbits(float f) {
  union { float f; uint32_t u; } v; v.f = f;
  uint32_t r = v.u + 0x7FFFu + ((v.u >> 16) & 1u);   // round-to-nearest-even
  return (uint16_t)(r >> 16);
}
static __device__ __forceinline__ __bf16 f2bf(float f) {
  uint16_t h = bfbits(f);
  return __builtin_bit_cast(__bf16, h);
}

static __device__ __forceinline__ float gelu_tanh(float v) {
  const float c = 0.7978845608028654f;
  float t = tanhf(c * (v + 0.044715f * v * v * v));
  return 0.5f * v * (1.0f + t);
}

// within-32 K-chunk block permutation: [0..7,16..23,8..15,24..31]
static __device__ __forceinline__ int kperm32(int r) {   // r in 0..31
  int b = r >> 3;
  int pb = ((b & 1) << 1) | (b >> 1);
  return (pb << 3) | (r & 7);
}

// ---------------------------------------------------------------------------
// 0) init: zero token-output region + gate-mean accumulators
// ---------------------------------------------------------------------------
__global__ void moe_init(float* __restrict__ out, float* __restrict__ me_sum) {
  size_t stride = (size_t)gridDim.x * blockDim.x;
  for (size_t i = (size_t)blockIdx.x * blockDim.x + threadIdx.x;
       i < (size_t)T_TOK * DMODEL; i += stride)
    out[i] = 0.f;
  if (blockIdx.x == 0 && threadIdx.x < NEXP) me_sum[threadIdx.x] = 0.f;
}

// ---------------------------------------------------------------------------
// 1) weight prepack: fp32 [E][K][N] -> bf16 fragment-tiled
//    [e][nt=n/16][kt=k/32][n0=n%16][hlf=lane half][16 k: kperm32(16*hlf + j)]
//    One 32B output chunk per thread.
// ---------------------------------------------------------------------------
template <int K, int N>
__global__ __launch_bounds__(256) void moe_prepack(const float* __restrict__ src,
                                                   __bf16* __restrict__ dst) {
  const int cid = blockIdx.x * 256 + threadIdx.x;
  int t = cid;
  const int hlf = t & 1;        t >>= 1;
  const int n0  = t & 15;       t >>= 4;
  const int kt  = t % (K / 32); t /= (K / 32);
  const int nt  = t % (N / 16); t /= (N / 16);
  const int e   = t;
  const int n   = nt * 16 + n0;
  const int kb  = kt * 32;
  const float* s = src + ((size_t)e * K + kb) * N + n;
  v16bf v;
#pragma unroll
  for (int j = 0; j < 16; ++j) v[j] = f2bf(s[(size_t)kperm32(hlf * 16 + j) * N]);
  *(v16bf*)(dst + (size_t)cid * 16) = v;
}

// ---------------------------------------------------------------------------
// 2) gating: one wave per token. logits = x @ wg, softmax, argmax.
// ---------------------------------------------------------------------------
__global__ __launch_bounds__(256) void moe_gate(
    const float* __restrict__ x, const float* __restrict__ wg,
    int* __restrict__ idx, float* __restrict__ gval,
    float* __restrict__ me_sum) {
  const int lane = threadIdx.x & 31;
  const int wid  = threadIdx.x >> 5;
  const int t    = blockIdx.x * 8 + wid;

  float acc[NEXP];
#pragma unroll
  for (int e = 0; e < NEXP; ++e) acc[e] = 0.f;

  const float* xr = x + (size_t)t * DMODEL;
  for (int d = lane; d < DMODEL; d += 32) {
    float xv = xr[d];
    const float* wr = wg + (size_t)d * NEXP;
#pragma unroll
    for (int e = 0; e < NEXP; ++e) acc[e] += xv * wr[e];
  }
#pragma unroll
  for (int e = 0; e < NEXP; ++e)
    for (int off = 16; off > 0; off >>= 1)
      acc[e] += __shfl_xor(acc[e], off, 32);

  float mx = acc[0]; int am = 0;
#pragma unroll
  for (int e = 1; e < NEXP; ++e)
    if (acc[e] > mx) { mx = acc[e]; am = e; }   // first-max tie-break, as argmax

  float ge[NEXP]; float s = 0.f;
#pragma unroll
  for (int e = 0; e < NEXP; ++e) { ge[e] = __expf(acc[e] - mx); s += ge[e]; }
  float inv = 1.f / s;

#pragma unroll
  for (int e = 0; e < NEXP; ++e)
    if (lane == e) atomicAdd(&me_sum[e], ge[e] * inv);
  if (lane == 0) { idx[t] = am; gval[t] = ge[am] * inv; }
}

// ---------------------------------------------------------------------------
// 3) scan: per-expert cumsum in token order, capacity drop, inverse map,
//    l_aux + exp_counts into d_out tail.
// ---------------------------------------------------------------------------
__global__ void moe_scan(const int* __restrict__ idx, const float* __restrict__ gval,
                         const float* __restrict__ me_sum,
                         int* __restrict__ token_of, float* __restrict__ gate_of,
                         float* __restrict__ dout) {
  __shared__ float part[NEXP];
  const int e = threadIdx.x;
  if (e < NEXP) {
    int cnt = 0;
    for (int t = 0; t < T_TOK; ++t) {
      if (idx[t] == e) {
        if (cnt < CAP) { token_of[e * CAP + cnt] = t; gate_of[e * CAP + cnt] = gval[t]; }
        ++cnt;
      }
    }
    int kept = cnt < CAP ? cnt : CAP;
    for (int c = kept; c < CAP; ++c) token_of[e * CAP + c] = -1;
    ((int*)dout)[(size_t)T_TOK * DMODEL + 1 + e] = cnt;   // exp_counts (pre-capacity)
    part[e] = (me_sum[e] / (float)T_TOK) * ((float)cnt / (float)T_TOK);
  }
  __syncthreads();
  if (e == 0) {
    float s = 0.f;
    for (int i = 0; i < NEXP; ++i) s += part[i];
    dout[(size_t)T_TOK * DMODEL] = s * (float)NEXP;       // l_aux
  }
}

// ---------------------------------------------------------------------------
// 4) dispatch: gather tokens -> [E, CAP, D] bf16, K-chunk permuted rows
// ---------------------------------------------------------------------------
__global__ __launch_bounds__(128) void moe_dispatch(
    const float* __restrict__ x, const int* __restrict__ token_of,
    __bf16* __restrict__ dispA) {
  const int row = blockIdx.x;                 // e*CAP + c
  const int t = token_of[row];
  __bf16* dst = dispA + (size_t)row * DMODEL;
  const __bf16 z = f2bf(0.f);
  if (t >= 0) {
    const float* src = x + (size_t)t * DMODEL;
    for (int d = threadIdx.x; d < DMODEL; d += 128) {
      int pos = (d & ~31) | kperm32(d & 31);
      dst[pos] = f2bf(src[d]);
    }
  } else {
    for (int d = threadIdx.x; d < DMODEL; d += 128) dst[d] = z;
  }
}

// ---------------------------------------------------------------------------
// Shared WMMA main loop: 128x128x32 block tile, 8 waves, wave tile 32x64.
// A: [128 rows][LDA] bf16 with K-chunk permutation pre-applied -> one 32B
//    ds read per fragment. B: fragment-tiled bf16 ([nt][K/32][512]) -> pure
//    b128 copies, zero conversion VALU in the loop.
// Register-pipelined: tile kt+1 global loads issue before tile kt's WMMAs;
// the loadcnt wait lands at the LDS store after the compute block.
// ---------------------------------------------------------------------------
template <int LDA, int K>
__device__ __forceinline__ void wmma_mainloop(
    const __bf16* __restrict__ A, const __bf16* __restrict__ Bt,
    __bf16 (&sA)[2][128 * 32], __bf16 (&sB)[2][128 * 32],
    v8f (&acc)[2][4]) {
  constexpr int NK = K / 32;
  const int tid  = threadIdx.x;
  const int lane = tid & 31, wid = tid >> 5;
  const int wm = wid >> 1, wn = wid & 1;
  const int lm = lane & 15, hlf = lane >> 4;

  const int bnt = tid >> 5, binner = tid & 31;   // B: 256 chunks of 32B

  v8bf  aReg[2];
  v16bf bReg;

  auto gload = [&](int kt) {
#pragma unroll
    for (int j = 0; j < 2; ++j) {                // A: 512 chunks of 16B
      int c = tid + j * 256;
      int row = c >> 2, q = c & 3;
      aReg[j] = *(const v8bf*)(A + (size_t)row * LDA + kt * 32 + q * 8);
    }
    bReg = *(const v16bf*)(Bt + ((size_t)bnt * NK + kt) * 512 + binner * 16);
  };
  auto sstore = [&](int buf) {
#pragma unroll
    for (int j = 0; j < 2; ++j) {
      int c = tid + j * 256;
      int row = c >> 2, q = c & 3;
      *(v8bf*)(&sA[buf][row * 32 + q * 8]) = aReg[j];
    }
    *(v16bf*)(&sB[buf][bnt * 512 + binner * 16]) = bReg;
  };

  gload(0);
  sstore(0);
  for (int kt = 0; kt < NK; ++kt) {
    __syncthreads();
    if (kt + 1 < NK) gload(kt + 1);              // issue loads, no wait yet
    if (kt + 2 < NK)
      __builtin_prefetch(Bt + ((size_t)bnt * NK + kt + 2) * 512 + binner * 16, 0, 0);

    const int buf = kt & 1;
    v16bf af[2], bfr[4];
#pragma unroll
    for (int mi = 0; mi < 2; ++mi)
      af[mi] = *(const v16bf*)(&sA[buf][(wm * 32 + mi * 16 + lm) * 32 + hlf * 16]);
#pragma unroll
    for (int ni = 0; ni < 4; ++ni)
      bfr[ni] = *(const v16bf*)(&sB[buf][(wn * 4 + ni) * 512 + lm * 32 + hlf * 16]);
#pragma unroll
    for (int mi = 0; mi < 2; ++mi)
#pragma unroll
      for (int ni = 0; ni < 4; ++ni)
        acc[mi][ni] = __builtin_amdgcn_wmma_f32_16x16x32_bf16(
            false, af[mi], false, bfr[ni], (short)0, acc[mi][ni], false, false);

    if (kt + 1 < NK) sstore((kt + 1) & 1);       // loadcnt wait lands here
  }
}

// ---------------------------------------------------------------------------
__global__ __launch_bounds__(256) void moe_gemm1(
    const __bf16* __restrict__ dispA, const __bf16* __restrict__ w1b,
    const float* __restrict__ b1, __bf16* __restrict__ hbuf) {
  const int e  = blockIdx.z;
  const int mb = blockIdx.y;     // 0..3   (CAP/128)
  const int nb = blockIdx.x;     // 0..31  (DFFN/128)
  const __bf16* A  = dispA + ((size_t)e * CAP + (size_t)mb * 128) * DMODEL;
  const __bf16* Bt = w1b + ((size_t)e * (DFFN / 16) + (size_t)nb * 8) * (DMODEL / 32) * 512;
  __bf16* Hg = hbuf + ((size_t)e * CAP + (size_t)mb * 128) * DFFN + (size_t)nb * 128;
  const float* bias = b1 + (size_t)e * DFFN + (size_t)nb * 128;

  __shared__ __align__(64) __bf16 sA[2][128 * 32];
  __shared__ __align__(64) __bf16 sB[2][128 * 32];

  const v8f vz = {0.f, 0.f, 0.f, 0.f, 0.f, 0.f, 0.f, 0.f};
  v8f acc[2][4];
#pragma unroll
  for (int i = 0; i < 2; ++i)
#pragma unroll
    for (int j = 0; j < 4; ++j) acc[i][j] = vz;

  wmma_mainloop<DMODEL, DMODEL>(A, Bt, sA, sB, acc);

  const int lane = threadIdx.x & 31, wid = threadIdx.x >> 5;
  const int wm = wid >> 1, wn = wid & 1;
  const int lm = lane & 15, hlf = lane >> 4;
#pragma unroll
  for (int mi = 0; mi < 2; ++mi) {
#pragma unroll
    for (int ni = 0; ni < 4; ++ni) {
      const int n = wn * 64 + ni * 16 + lm;
      // h is GEMM2's A: store with its K-chunk permutation pre-applied
      const int np = (n & ~31) | kperm32(n & 31);
      const float bv = bias[n];
#pragma unroll
      for (int r = 0; r < 8; ++r) {
        int m = wm * 32 + mi * 16 + hlf * 8 + r;
        Hg[(size_t)m * DFFN + np] = f2bf(gelu_tanh(acc[mi][ni][r] + bv));
      }
    }
  }
}

__global__ __launch_bounds__(256) void moe_gemm2(
    const __bf16* __restrict__ hbuf, const __bf16* __restrict__ w2b,
    const float* __restrict__ b2, const int* __restrict__ token_of,
    const float* __restrict__ gate_of, float* __restrict__ out) {
  const int e  = blockIdx.z;
  const int mb = blockIdx.y;     // 0..3  (CAP/128)
  const int nb = blockIdx.x;     // 0..7  (DMODEL/128)
  const __bf16* A  = hbuf + ((size_t)e * CAP + (size_t)mb * 128) * DFFN;
  const __bf16* Bt = w2b + ((size_t)e * (DMODEL / 16) + (size_t)nb * 8) * (DFFN / 32) * 512;
  const float* bias = b2 + (size_t)e * DMODEL + (size_t)nb * 128;

  __shared__ __align__(64) __bf16 sA[2][128 * 32];
  __shared__ __align__(64) __bf16 sB[2][128 * 32];

  const v8f vz = {0.f, 0.f, 0.f, 0.f, 0.f, 0.f, 0.f, 0.f};
  v8f acc[2][4];
#pragma unroll
  for (int i = 0; i < 2; ++i)
#pragma unroll
    for (int j = 0; j < 4; ++j) acc[i][j] = vz;

  wmma_mainloop<DFFN, DFFN>(A, Bt, sA, sB, acc);

  const int lane = threadIdx.x & 31, wid = threadIdx.x >> 5;
  const int wm = wid >> 1, wn = wid & 1;
  const int lm = lane & 15, hlf = lane >> 4;
#pragma unroll
  for (int mi = 0; mi < 2; ++mi) {
    int tok[8]; float gv[8];
#pragma unroll
    for (int r = 0; r < 8; ++r) {
      int m = mb * 128 + wm * 32 + mi * 16 + hlf * 8 + r;
      tok[r] = token_of[e * CAP + m];
      gv[r]  = gate_of[e * CAP + m];
    }
#pragma unroll
    for (int ni = 0; ni < 4; ++ni) {
      const int n = wn * 64 + ni * 16 + lm;
      const float bv = bias[n];
      const int dcol = nb * 128 + n;
#pragma unroll
      for (int r = 0; r < 8; ++r)
        if (tok[r] >= 0)
          out[(size_t)tok[r] * DMODEL + dcol] = gv[r] * (acc[mi][ni][r] + bv);
    }
  }
}

// ---------------------------------------------------------------------------
extern "C" void kernel_launch(void* const* d_in, const int* in_sizes, int n_in,
                              void* d_out, int out_size, void* d_ws, size_t ws_size,
                              hipStream_t stream) {
  (void)in_sizes; (void)n_in; (void)out_size; (void)ws_size;
  const float* x  = (const float*)d_in[0];   // [T, D]
  const float* wg = (const float*)d_in[1];   // [D, E]
  const float* w1 = (const float*)d_in[2];   // [E, D, DFF]
  const float* b1 = (const float*)d_in[3];   // [E, DFF]
  const float* w2 = (const float*)d_in[4];   // [E, DFF, D]
  const float* b2 = (const float*)d_in[5];   // [E, D]
  float* out = (float*)d_out;                // [T*D] out, [1] l_aux, [E] exp_counts

  // workspace layout (all offsets multiples of 256 bytes)
  char* w = (char*)d_ws;
  int*    idx      = (int*)w;      w += (size_t)T_TOK * 4;
  float*  gval     = (float*)w;    w += (size_t)T_TOK * 4;
  float*  me_sum   = (float*)w;    w += 256;
  int*    token_of = (int*)w;      w += (size_t)NEXP * CAP * 4;
  float*  gate_of  = (float*)w;    w += (size_t)NEXP * CAP * 4;
  __bf16* dispA    = (__bf16*)w;   w += (size_t)NEXP * CAP * DMODEL * 2;    // 8 MB
  __bf16* hbuf     = (__bf16*)w;   w += (size_t)NEXP * CAP * DFFN * 2;      // 32 MB
  __bf16* w1b      = (__bf16*)w;   w += (size_t)NEXP * DMODEL * DFFN * 2;   // 64 MB
  __bf16* w2b      = (__bf16*)w;                                            // 64 MB

  // one-time weight prepack (fp32 -> bf16, WMMA fragment layout)
  moe_prepack<DMODEL, DFFN><<<8192, 256, 0, stream>>>(w1, w1b);
  moe_prepack<DFFN, DMODEL><<<8192, 256, 0, stream>>>(w2, w2b);

  moe_init<<<4096, 256, 0, stream>>>(out, me_sum);
  moe_gate<<<T_TOK / 8, 256, 0, stream>>>(x, wg, idx, gval, me_sum);
  moe_scan<<<1, 32, 0, stream>>>(idx, gval, me_sum, token_of, gate_of, out);
  moe_dispatch<<<NEXP * CAP, 128, 0, stream>>>(x, token_of, dispA);

  dim3 g1(DFFN / 128, CAP / 128, NEXP);   // (32, 4, 8)
  moe_gemm1<<<g1, 256, 0, stream>>>(dispA, w1b, b1, hbuf);
  dim3 g2(DMODEL / 128, CAP / 128, NEXP); // (8, 4, 8)
  moe_gemm2<<<g2, 256, 0, stream>>>(hbuf, w2b, b2, token_of, gate_of, out);
}
